// RBFFeatureExtractor_65481071409526
// MI455X (gfx1250) — matrix operand order repaired and checked by reading, not compiled, not run on hardware
//
#include <hip/hip_runtime.h>
#include <hip/hip_bf16.h>

typedef __attribute__((ext_vector_type(16))) _Float16 v16h;
typedef __attribute__((ext_vector_type(8)))  _Float16 v8h;
typedef __attribute__((ext_vector_type(8)))  float    v8f;

#define N_ROWS 16384   // rows of x
#define M_ROWS 4096    // rows of support (= output cols)
#define DDIM   256     // feature dim
#define BM     128     // block tile M (x rows)
#define BN     128     // block tile N (support rows)
#define BK     32      // K step (f16 WMMA depth)
#define LP     40      // LDS row pitch in halfs (32 data + 8 pad -> 80B, spreads banks)
#define KTILES (DDIM / BK)

// ---------------------------------------------------------------------------
// Fused pre-pass: f32 -> f16 copy of a row + its squared L2 norm (f32).
// One wave32 per row of 256 floats.
// ---------------------------------------------------------------------------
__global__ __launch_bounds__(256) void prep_kernel(const float* __restrict__ src,
                                                   _Float16* __restrict__ hdst,
                                                   float* __restrict__ ndst) {
    const int row  = blockIdx.x * 8 + (threadIdx.x >> 5);
    const int lane = threadIdx.x & 31;
    const float* p = src + (size_t)row * DDIM + lane * 8;
    float4 a = ((const float4*)p)[0];
    float4 b = ((const float4*)p)[1];
    float s = a.x*a.x + a.y*a.y + a.z*a.z + a.w*a.w
            + b.x*b.x + b.y*b.y + b.z*b.z + b.w*b.w;
    v8h h = { (_Float16)a.x, (_Float16)a.y, (_Float16)a.z, (_Float16)a.w,
              (_Float16)b.x, (_Float16)b.y, (_Float16)b.z, (_Float16)b.w };
    *(v8h*)(hdst + (size_t)row * DDIM + lane * 8) = h;
#pragma unroll
    for (int off = 16; off > 0; off >>= 1) s += __shfl_xor(s, off, 32);
    if (lane == 0) ndst[row] = s;
}

// Fallback-path pre-pass (norms only, inputs stay f32)
__global__ __launch_bounds__(256) void row_norms_kernel(const float* __restrict__ src,
                                                        float* __restrict__ dst) {
    const int row  = blockIdx.x * 8 + (threadIdx.x >> 5);
    const int lane = threadIdx.x & 31;
    const float4* p = (const float4*)(src + (size_t)row * DDIM) + lane * 2;
    float4 a = p[0];
    float4 b = p[1];
    float s = a.x*a.x + a.y*a.y + a.z*a.z + a.w*a.w
            + b.x*b.x + b.y*b.y + b.z*b.z + b.w*b.w;
#pragma unroll
    for (int off = 16; off > 0; off >>= 1) s += __shfl_xor(s, off, 32);
    if (lane == 0) dst[row] = s;
}

// ---------------------------------------------------------------------------
// Fragment helpers (CDNA5 WMMA VGPR layouts, CDNA5 ISA)
// ---------------------------------------------------------------------------
__device__ __forceinline__ v16h combine16(v8h lo, v8h hi) {
    v16h f;
#pragma unroll
    for (int i = 0; i < 8; ++i) { f[i] = lo[i]; f[i + 8] = hi[i]; }
    return f;
}

// A 16x32 f16: lane holds row l&15; contiguous K run of 16 starting at (l>=16)*16
__device__ __forceinline__ v16h load_frag_a(const _Float16* base) {
    v8h lo = *(const v8h*)(base);
    v8h hi = *(const v8h*)(base + 8);
    return combine16(lo, hi);
}

// B 32x16 f16: lane holds col l&15; contiguous K run of 16 starting at (l>=16)*16
__device__ __forceinline__ v16h load_frag_b(const _Float16* base) {
    v8h lo = *(const v8h*)(base);
    v8h hi = *(const v8h*)(base + 8);
    return combine16(lo, hi);
}

// 32B async DMA global->LDS (two b128; inst offset applies to BOTH addresses)
__device__ __forceinline__ void async_copy32(unsigned lds_byte_addr,
                                             const _Float16* src) {
    asm volatile("global_load_async_to_lds_b128 %0, %1, off\n\t"
                 "global_load_async_to_lds_b128 %0, %1, off offset:16"
                 :: "v"(lds_byte_addr), "v"(src) : "memory");
}

__device__ __forceinline__ void wait_async0() {
    asm volatile("s_wait_asynccnt 0x0" ::: "memory");
}

// ---------------------------------------------------------------------------
// Primary kernel: f16 inputs, double-buffered LDS, async-DMA staging.
// 128x128 tile / 256 threads; 8 waves as 2(M) x 4(N); wave = 64x32 via 4x2
// 16x16 WMMA accumulators.
// ---------------------------------------------------------------------------
__global__ __launch_bounds__(256) void rbf_wmma_async_kernel(
        const _Float16* __restrict__ XH, const _Float16* __restrict__ SH,
        const float* __restrict__ xsq,  const float* __restrict__ ssq,
        float* __restrict__ out) {
    __shared__ __align__(16) _Float16 As[2][BM * LP];
    __shared__ __align__(16) _Float16 Bs[2][BN * LP];

    const int tid  = threadIdx.x;
    const int lane = tid & 31;
    const int wave = tid >> 5;

    const int bm = blockIdx.y * BM;
    const int bn = blockIdx.x * BN;

    const int wm = (wave >> 2) * 64;
    const int wn = (wave & 3) * 32;

    // staging coordinates: each thread moves 16 halfs (32B) of a 32-half row
    const int srow = tid >> 1;
    const int scol = (tid & 1) * 16;

    // fragment lane decomposition
    const int fr  = lane & 15;
    const int hi  = lane >> 4;
    const int fka = hi * 16;
    const int fkb = hi * 16;

    const _Float16* xg = XH + (size_t)(bm + srow) * DDIM + scol;
    const _Float16* sg = SH + (size_t)(bn + srow) * DDIM + scol;

    // LDS byte addresses for this thread's staging slot (low 32 bits of the
    // generic pointer to __shared__ == LDS byte offset)
    const unsigned aoff = srow * (LP * 2) + scol * 2;
    const unsigned a0 = (unsigned)(uintptr_t)&As[0][0] + aoff;
    const unsigned a1 = (unsigned)(uintptr_t)&As[1][0] + aoff;
    const unsigned b0 = (unsigned)(uintptr_t)&Bs[0][0] + aoff;
    const unsigned b1 = (unsigned)(uintptr_t)&Bs[1][0] + aoff;

    v8f acc[4][2] = {};

    // prologue: stage k-tile 0 into buffer 0
    async_copy32(a0, xg);
    async_copy32(b0, sg);
    wait_async0();
    __syncthreads();

    for (int kt = 0; kt < KTILES; ++kt) {
        const int cur = kt & 1;
        const _Float16* Ab = &As[cur][0];
        const _Float16* Bb = &Bs[cur][0];

        // prefetch next k-tile into the other buffer (async DMA, overlaps WMMA)
        if (kt + 1 < KTILES) {
            const int k1 = (kt + 1) * BK;
            async_copy32(cur ? a0 : a1, xg + k1);
            async_copy32(cur ? b0 : b1, sg + k1);
        }

        v16h afrag[4];
        v16h bfrag[2];
#pragma unroll
        for (int i = 0; i < 4; ++i)
            afrag[i] = load_frag_a(&Ab[(wm + i * 16 + fr) * LP + fka]);
#pragma unroll
        for (int j = 0; j < 2; ++j)
            bfrag[j] = load_frag_b(&Bb[(wn + j * 16 + fr) * LP + fkb]);

#pragma unroll
        for (int i = 0; i < 4; ++i)
#pragma unroll
            for (int j = 0; j < 2; ++j)
                acc[i][j] = __builtin_amdgcn_wmma_f32_16x16x32_f16(
                    false, afrag[i], false, bfrag[j],
                    (short)0, acc[i][j], false, false);

        wait_async0();     // this wave's DMA into next buffer has landed
        __syncthreads();   // everyone's DMA landed; reads of 'cur' are done
    }

    // Epilogue: exp(-(max(xsq + ssq - 2*dot, 0))), GAMMA == 1
#pragma unroll
    for (int i = 0; i < 4; ++i) {
        const int r0 = bm + wm + i * 16 + hi * 8;
        float xs[8];
#pragma unroll
        for (int v = 0; v < 8; ++v) xs[v] = xsq[r0 + v];
#pragma unroll
        for (int j = 0; j < 2; ++j) {
            const int c = bn + wn + j * 16 + fr;
            const float sv = ssq[c];
            float* o = out + (size_t)r0 * M_ROWS + c;
#pragma unroll
            for (int v = 0; v < 8; ++v) {
                float d2 = xs[v] + sv - 2.0f * acc[i][j][v];
                d2 = fmaxf(d2, 0.0f);
                o[(size_t)v * M_ROWS] = __expf(-d2);
            }
        }
    }
}

// ---------------------------------------------------------------------------
// Fallback kernel: f32 inputs, convert-on-the-fly.
// Used only if d_ws is too small for f16 copies.
// ---------------------------------------------------------------------------
__device__ __forceinline__ void cvt_stage16(_Float16* __restrict__ dst,
                                            const float* __restrict__ g) {
    const float4* gp = (const float4*)g;
    float4 a = gp[0], b = gp[1], c = gp[2], d = gp[3];
    v8h h0 = { (_Float16)a.x, (_Float16)a.y, (_Float16)a.z, (_Float16)a.w,
               (_Float16)b.x, (_Float16)b.y, (_Float16)b.z, (_Float16)b.w };
    v8h h1 = { (_Float16)c.x, (_Float16)c.y, (_Float16)c.z, (_Float16)c.w,
               (_Float16)d.x, (_Float16)d.y, (_Float16)d.z, (_Float16)d.w };
    *(v8h*)dst       = h0;
    *(v8h*)(dst + 8) = h1;
}

__global__ __launch_bounds__(256) void rbf_wmma_kernel(const float* __restrict__ X,
                                                       const float* __restrict__ S,
                                                       const float* __restrict__ xsq,
                                                       const float* __restrict__ ssq,
                                                       float* __restrict__ out) {
    __shared__ __align__(16) _Float16 As[BM * LP];
    __shared__ __align__(16) _Float16 Bs[BN * LP];

    const int tid  = threadIdx.x;
    const int lane = tid & 31;
    const int wave = tid >> 5;
    const int bm = blockIdx.y * BM;
    const int bn = blockIdx.x * BN;
    const int wm = (wave >> 2) * 64;
    const int wn = (wave & 3) * 32;
    const int srow = tid >> 1;
    const int scol = (tid & 1) * 16;
    const int fr  = lane & 15;
    const int hi  = lane >> 4;
    const int fka = hi * 16;
    const int fkb = hi * 16;

    v8f acc[4][2] = {};

    const float* xg = X + (size_t)(bm + srow) * DDIM + scol;
    const float* sg = S + (size_t)(bn + srow) * DDIM + scol;
    _Float16* asd = &As[srow * LP + scol];
    _Float16* bsd = &Bs[srow * LP + scol];

    for (int k0 = 0; k0 < DDIM; k0 += BK) {
        cvt_stage16(asd, xg + k0);
        cvt_stage16(bsd, sg + k0);
        __syncthreads();

        v16h afrag[4];
        v16h bfrag[2];
#pragma unroll
        for (int i = 0; i < 4; ++i)
            afrag[i] = load_frag_a(&As[(wm + i * 16 + fr) * LP + fka]);
#pragma unroll
        for (int j = 0; j < 2; ++j)
            bfrag[j] = load_frag_b(&Bs[(wn + j * 16 + fr) * LP + fkb]);

#pragma unroll
        for (int i = 0; i < 4; ++i)
#pragma unroll
            for (int j = 0; j < 2; ++j)
                acc[i][j] = __builtin_amdgcn_wmma_f32_16x16x32_f16(
                    false, afrag[i], false, bfrag[j],
                    (short)0, acc[i][j], false, false);

        __syncthreads();
    }

#pragma unroll
    for (int i = 0; i < 4; ++i) {
        const int r0 = bm + wm + i * 16 + hi * 8;
        float xs[8];
#pragma unroll
        for (int v = 0; v < 8; ++v) xs[v] = xsq[r0 + v];
#pragma unroll
        for (int j = 0; j < 2; ++j) {
            const int c = bn + wn + j * 16 + fr;
            const float sv = ssq[c];
            float* o = out + (size_t)r0 * M_ROWS + c;
#pragma unroll
            for (int v = 0; v < 8; ++v) {
                float d2 = xs[v] + sv - 2.0f * acc[i][j][v];
                d2 = fmaxf(d2, 0.0f);
                o[(size_t)v * M_ROWS] = __expf(-d2);
            }
        }
    }
}

// ---------------------------------------------------------------------------
extern "C" void kernel_launch(void* const* d_in, const int* in_sizes, int n_in,
                              void* d_out, int out_size, void* d_ws, size_t ws_size,
                              hipStream_t stream) {
    const float* x = (const float*)d_in[0];
    const float* s = (const float*)d_in[1];
    float* out = (float*)d_out;

    dim3 grid(M_ROWS / BN, N_ROWS / BM);   // (32, 128)

    const size_t f16_bytes = ((size_t)N_ROWS + M_ROWS) * DDIM * sizeof(_Float16);
    const size_t norm_bytes = ((size_t)N_ROWS + M_ROWS) * sizeof(float);

    if (ws_size >= f16_bytes + norm_bytes) {
        // primary path: f16 copies + norms in d_ws, async-DMA GEMM
        _Float16* xh = (_Float16*)d_ws;
        _Float16* sh = xh + (size_t)N_ROWS * DDIM;
        float* xsq = (float*)(sh + (size_t)M_ROWS * DDIM);
        float* ssq = xsq + N_ROWS;

        prep_kernel<<<N_ROWS / 8, 256, 0, stream>>>(x, xh, xsq);
        prep_kernel<<<M_ROWS / 8, 256, 0, stream>>>(s, sh, ssq);
        rbf_wmma_async_kernel<<<grid, 256, 0, stream>>>(xh, sh, xsq, ssq, out);
    } else {
        // fallback: norms only in d_ws, convert f32->f16 on the fly
        float* xsq = (float*)d_ws;
        float* ssq = xsq + N_ROWS;

        row_norms_kernel<<<N_ROWS / 8, 256, 0, stream>>>(x, xsq);
        row_norms_kernel<<<M_ROWS / 8, 256, 0, stream>>>(s, ssq);
        rbf_wmma_kernel<<<grid, 256, 0, stream>>>(x, s, xsq, ssq, out);
    }
}
